// SelfAttention_4535485464962
// MI455X (gfx1250) — matrix operand order repaired and checked by reading, not compiled, not run on hardware
//
#include <hip/hip_runtime.h>
#include <hip/hip_bf16.h>

// ---------------------------------------------------------------------------
// Causal multi-head self-attention forward, CDNA5 (gfx1250) WMMA implementation
//   B=2, T=2048, C=1024, H=16, HD=64
// All matmuls run on v_wmma_f32_16x16x32_f16 (fp16 in, fp32 accumulate).
// ---------------------------------------------------------------------------

typedef __attribute__((ext_vector_type(16))) _Float16 v16h;
typedef __attribute__((ext_vector_type(8)))  _Float16 v8h;
typedef __attribute__((ext_vector_type(8)))  float    v8f;

constexpr int Bb   = 2;
constexpr int Tt   = 2048;
constexpr int Cc   = 1024;
constexpr int Hh   = 16;
constexpr int HD   = 64;
constexpr int Mm   = Bb * Tt;      // 4096 rows
constexpr int Nqkv = 3 * Cc;       // 3072

// ---- workspace layout (bytes). Requires ~48 MiB of d_ws. -------------------
constexpr size_t XH_OFF    = 0;                       // x fp16        (8 MiB)
constexpr size_t WQKVT_OFF = XH_OFF    + (size_t)Mm * Cc * 2;      // 6 MiB
constexpr size_t QH_OFF    = WQKVT_OFF + (size_t)Cc * Nqkv * 2;    // 8 MiB
constexpr size_t KH_OFF    = QH_OFF    + (size_t)Bb * Hh * Tt * HD * 2;
constexpr size_t VT_OFF    = KH_OFF    + (size_t)Bb * Hh * Tt * HD * 2;
constexpr size_t AH_OFF    = VT_OFF    + (size_t)Bb * Hh * Tt * HD * 2;
constexpr size_t WPT_OFF   = AH_OFF    + (size_t)Mm * Cc * 2;      // 2 MiB

// ---------------------------------------------------------------------------
// Fragment helpers (ISA 7.12.2 layouts, wave32)
// ---------------------------------------------------------------------------
__device__ __forceinline__ v16h combine16(v8h lo, v8h hi) {
  v16h r;
#pragma unroll
  for (int i = 0; i < 8; ++i) { r[i] = lo[i]; r[i + 8] = hi[i]; }
  return r;
}

// A-matrix 16x32 f16 fragment: lane holds row (lane&15); lanes 0-15 carry
// K = 0..15, lanes 16-31 carry K = 16..31.
__device__ __forceinline__ v16h load_a_frag(const _Float16* __restrict__ base,
                                            int ld, int row0, int k0, int lane) {
  const int row = row0 + (lane & 15);
  const int g   = lane >> 4;
  const _Float16* p = base + (size_t)row * ld + k0 + g * 16;
  v8h lo = *(const v8h*)(p);
  v8h hi = *(const v8h*)(p + 8);
  return combine16(lo, hi);
}

// B-matrix 32x16 f16 fragment from a column-major store (ld = column stride):
// lane holds column (lane&15), K-range (lane>>4)*16 .. +15 contiguous.
__device__ __forceinline__ v16h load_b_frag(const _Float16* __restrict__ base,
                                            int ld, int col0, int k0, int lane) {
  const int col = col0 + (lane & 15);
  const int g   = lane >> 4;
  return *(const v16h*)(base + (size_t)col * ld + k0 + g * 16);
}

__device__ __forceinline__ v8f wmma16(v16h a, v16h b, v8f c) {
  // D = A(16x32) * B(32x16) + C, fp32 accumulate
  return __builtin_amdgcn_wmma_f32_16x16x32_f16(false, a, false, b,
                                                (short)0, c, false, false);
}

// ---------------------------------------------------------------------------
// Conversion kernels
// ---------------------------------------------------------------------------
__global__ void cvt_f32_to_f16_kernel(const float* __restrict__ src,
                                      _Float16* __restrict__ dst, int n) {
  int i = blockIdx.x * blockDim.x + threadIdx.x;
  if (i < n) dst[i] = (_Float16)src[i];
}

// src is (K x N) fp32 row-major; dst is (N x K) fp16 (i.e., column-major src)
__global__ void transpose_to_f16_kernel(const float* __restrict__ src,
                                        _Float16* __restrict__ dst, int K, int N) {
  int i = blockIdx.x * blockDim.x + threadIdx.x;
  if (i < K * N) {
    int k = i / N;
    int n = i - k * N;
    dst[(size_t)n * K + k] = (_Float16)src[i];
  }
}

// ---------------------------------------------------------------------------
// QKV GEMM: (4096 x 1024) x (1024 x 3072) + bias, scatter into Q/K/V^T heads
// Block = 256 threads (8 waves); wave tile = 32x64 (2x4 WMMA tiles)
// ---------------------------------------------------------------------------
__global__ __launch_bounds__(256)
void gemm_qkv_kernel(const _Float16* __restrict__ Xh,
                     const _Float16* __restrict__ Wt,   // (3072 x 1024) col-major of W
                     const float*    __restrict__ bias,
                     _Float16* __restrict__ Qh,         // [b][h][t][hd]
                     _Float16* __restrict__ Kh,         // [b][h][t][hd]
                     _Float16* __restrict__ Vt) {       // [b][h][hd][t]
  const int lane = threadIdx.x & 31;
  const int wave = threadIdx.x >> 5;
  const int m0 = blockIdx.y * 64  + (wave >> 2) * 32;
  const int n0 = blockIdx.x * 256 + (wave & 3) * 64;

  v8f acc[2][4];
#pragma unroll
  for (int i = 0; i < 2; ++i)
#pragma unroll
    for (int j = 0; j < 4; ++j) acc[i][j] = (v8f){0, 0, 0, 0, 0, 0, 0, 0};

  for (int k0 = 0; k0 < Cc; k0 += 32) {
    if (k0 + 32 < Cc) {   // lowers to global_prefetch_b8
      __builtin_prefetch(Xh + (size_t)(m0 + (lane & 15)) * Cc + k0 + 32, 0, 0);
      __builtin_prefetch(Wt + (size_t)(n0 + (lane & 15)) * Cc + k0 + 32, 0, 0);
    }
    v16h a0 = load_a_frag(Xh, Cc, m0,      k0, lane);
    v16h a1 = load_a_frag(Xh, Cc, m0 + 16, k0, lane);
#pragma unroll
    for (int j = 0; j < 4; ++j) {
      v16h bf = load_b_frag(Wt, Cc, n0 + j * 16, k0, lane);
      acc[0][j] = wmma16(a0, bf, acc[0][j]);
      acc[1][j] = wmma16(a1, bf, acc[1][j]);
    }
  }

  const int g = lane >> 4, col = lane & 15;
#pragma unroll
  for (int i = 0; i < 2; ++i)
#pragma unroll
    for (int j = 0; j < 4; ++j)
#pragma unroll
      for (int r = 0; r < 8; ++r) {
        const int m = m0 + i * 16 + r + 8 * g;
        const int n = n0 + j * 16 + col;
        const _Float16 hv = (_Float16)(acc[i][j][r] + bias[n]);
        const int which = n >> 10;          // 0=q 1=k 2=v
        const int hh    = (n & 1023) >> 6;  // head
        const int hd    = n & 63;
        const int bi    = m >> 11;          // batch (T = 2048)
        const int t     = m & 2047;
        const size_t bht = ((size_t)(bi * Hh + hh)) * Tt;
        if (which == 0)      Qh[(bht + t) * HD + hd] = hv;
        else if (which == 1) Kh[(bht + t) * HD + hd] = hv;
        else                 Vt[(((size_t)(bi * Hh + hh)) * HD + hd) * Tt + t] = hv;
      }
}

// ---------------------------------------------------------------------------
// Flash attention: one wave per (b,h, 16-row q tile); streams K/V in 32-chunks
// with an online softmax. S and O products via v_wmma_f32_16x16x32_f16.
// ---------------------------------------------------------------------------
__global__ __launch_bounds__(32)
void attn_kernel(const _Float16* __restrict__ Qh,
                 const _Float16* __restrict__ Kh,
                 const _Float16* __restrict__ Vt,
                 _Float16* __restrict__ Ah) {            // [b][t][h*64+hd] fp16
  __shared__ _Float16 Plds[16 * 32];

  const int lane = threadIdx.x & 31;
  const int qt   = blockIdx.x;       // q tile (16 rows)
  const int bh   = blockIdx.y;       // b*H + h
  const int bi   = bh >> 4, h = bh & 15;
  const int q0   = qt * 16;
  const int g    = lane >> 4, col = lane & 15;

  const _Float16* Qp = Qh + (size_t)bh * Tt * HD;
  const _Float16* Kp = Kh + (size_t)bh * Tt * HD;
  const _Float16* Vp = Vt + (size_t)bh * HD * Tt;

  v16h qf0 = load_a_frag(Qp, HD, q0, 0,  lane);
  v16h qf1 = load_a_frag(Qp, HD, q0, 32, lane);

  v8f o[4];
#pragma unroll
  for (int j = 0; j < 4; ++j) o[j] = (v8f){0, 0, 0, 0, 0, 0, 0, 0};
  float mi[8], li[8];
#pragma unroll
  for (int r = 0; r < 8; ++r) { mi[r] = -3.0e38f; li[r] = 0.0f; }

  const float scale = 0.125f;        // 1/sqrt(64)
  const int kend = q0 + 16;          // causal bound (exclusive)

  for (int kb = 0; kb < kend; kb += 32) {
    // ---- S = Q * K^T for 16x32 chunk (two 16x16 tiles) --------------------
    v8f s0 = (v8f){0, 0, 0, 0, 0, 0, 0, 0};
    v8f s1 = (v8f){0, 0, 0, 0, 0, 0, 0, 0};
    {
      v16h b0 = load_b_frag(Kp, HD, kb, 0,  lane);
      v16h b1 = load_b_frag(Kp, HD, kb, 32, lane);
      s0 = wmma16(qf0, b0, s0);
      s0 = wmma16(qf1, b1, s0);
    }
    {
      v16h b0 = load_b_frag(Kp, HD, kb + 16, 0,  lane);
      v16h b1 = load_b_frag(Kp, HD, kb + 16, 32, lane);
      s1 = wmma16(qf0, b0, s1);
      s1 = wmma16(qf1, b1, s1);
    }

    // ---- online softmax (rows live across the 16-lane half groups) --------
    float p0[8], p1[8], corr[8];
#pragma unroll
    for (int r = 0; r < 8; ++r) {
      const int qrow = q0 + r + 8 * g;
      float v0 = s0[r] * scale; if (kb + col      > qrow) v0 = -3.0e38f;
      float v1 = s1[r] * scale; if (kb + 16 + col > qrow) v1 = -3.0e38f;
      float rm = fmaxf(v0, v1);
#pragma unroll
      for (int off = 1; off < 16; off <<= 1)
        rm = fmaxf(rm, __shfl_xor(rm, off, 32));
      const float mnew = fmaxf(mi[r], rm);
      const float c    = __expf(mi[r] - mnew);
      v0 = __expf(v0 - mnew);
      v1 = __expf(v1 - mnew);
      float rs = v0 + v1;
#pragma unroll
      for (int off = 1; off < 16; off <<= 1)
        rs += __shfl_xor(rs, off, 32);
      li[r] = li[r] * c + rs;
      mi[r] = mnew;
      corr[r] = c; p0[r] = v0; p1[r] = v1;
    }
#pragma unroll
    for (int j = 0; j < 4; ++j)
#pragma unroll
      for (int r = 0; r < 8; ++r) o[j][r] *= corr[r];

    // ---- reshape P: C-layout -> A-fragment via LDS ------------------------
#pragma unroll
    for (int r = 0; r < 8; ++r) {
      const int mrow = r + 8 * g;
      Plds[mrow * 32 + col]      = (_Float16)p0[r];
      Plds[mrow * 32 + 16 + col] = (_Float16)p1[r];
    }
    __syncthreads();
    v16h pf;
    {
      const _Float16* pr = &Plds[(lane & 15) * 32 + g * 16];
      v8h lo = *(const v8h*)(pr);
      v8h hi = *(const v8h*)(pr + 8);
      pf = combine16(lo, hi);
    }
    // ---- O += P * V (V^T stored [hd][t], so columns are contiguous) -------
#pragma unroll
    for (int j = 0; j < 4; ++j) {
      v16h bv = load_b_frag(Vp, Tt, j * 16, kb, lane);
      o[j] = wmma16(pf, bv, o[j]);
    }
    __syncthreads();
  }

  // ---- normalize + write attention output (fp16, row-major B*T x C) -------
#pragma unroll
  for (int r = 0; r < 8; ++r) {
    const float inv = 1.0f / li[r];
    const int t = q0 + r + 8 * g;
#pragma unroll
    for (int j = 0; j < 4; ++j) {
      const int c = h * HD + j * 16 + col;
      Ah[((size_t)(bi * Tt + t)) * Cc + c] = (_Float16)(o[j][r] * inv);
    }
  }
}

// ---------------------------------------------------------------------------
// Output projection: (4096 x 1024) x (1024 x 1024) + bias -> fp32
// ---------------------------------------------------------------------------
__global__ __launch_bounds__(256)
void gemm_proj_kernel(const _Float16* __restrict__ Ah,
                      const _Float16* __restrict__ Wpt,   // (1024 x 1024) col-major
                      const float*    __restrict__ bias,
                      float* __restrict__ out) {
  const int lane = threadIdx.x & 31;
  const int wave = threadIdx.x >> 5;
  const int m0 = blockIdx.y * 64  + (wave >> 2) * 32;
  const int n0 = blockIdx.x * 256 + (wave & 3) * 64;

  v8f acc[2][4];
#pragma unroll
  for (int i = 0; i < 2; ++i)
#pragma unroll
    for (int j = 0; j < 4; ++j) acc[i][j] = (v8f){0, 0, 0, 0, 0, 0, 0, 0};

  for (int k0 = 0; k0 < Cc; k0 += 32) {
    if (k0 + 32 < Cc) {
      __builtin_prefetch(Ah  + (size_t)(m0 + (lane & 15)) * Cc + k0 + 32, 0, 0);
      __builtin_prefetch(Wpt + (size_t)(n0 + (lane & 15)) * Cc + k0 + 32, 0, 0);
    }
    v16h a0 = load_a_frag(Ah, Cc, m0,      k0, lane);
    v16h a1 = load_a_frag(Ah, Cc, m0 + 16, k0, lane);
#pragma unroll
    for (int j = 0; j < 4; ++j) {
      v16h bf = load_b_frag(Wpt, Cc, n0 + j * 16, k0, lane);
      acc[0][j] = wmma16(a0, bf, acc[0][j]);
      acc[1][j] = wmma16(a1, bf, acc[1][j]);
    }
  }

  const int g = lane >> 4, col = lane & 15;
#pragma unroll
  for (int i = 0; i < 2; ++i)
#pragma unroll
    for (int j = 0; j < 4; ++j)
#pragma unroll
      for (int r = 0; r < 8; ++r) {
        const int m = m0 + i * 16 + r + 8 * g;
        const int n = n0 + j * 16 + col;
        out[(size_t)m * Cc + n] = acc[i][j][r] + bias[n];
      }
}

// ---------------------------------------------------------------------------
// kernel_launch
// inputs: 0=x(f32) 1=mask(bool, ignored: causal is known) 2=W_qkv 3=b_qkv
//         4=W_proj 5=b_proj ; output: f32 (B*T*C)
// ---------------------------------------------------------------------------
extern "C" void kernel_launch(void* const* d_in, const int* in_sizes, int n_in,
                              void* d_out, int out_size, void* d_ws, size_t ws_size,
                              hipStream_t stream) {
  (void)in_sizes; (void)n_in; (void)out_size; (void)ws_size;

  const float* x     = (const float*)d_in[0];
  const float* Wqkv  = (const float*)d_in[2];
  const float* bqkv  = (const float*)d_in[3];
  const float* Wproj = (const float*)d_in[4];
  const float* bproj = (const float*)d_in[5];
  float* out = (float*)d_out;

  char* ws = (char*)d_ws;
  _Float16* Xh    = (_Float16*)(ws + XH_OFF);
  _Float16* Wqkvt = (_Float16*)(ws + WQKVT_OFF);
  _Float16* Qh    = (_Float16*)(ws + QH_OFF);
  _Float16* Kh    = (_Float16*)(ws + KH_OFF);
  _Float16* Vt    = (_Float16*)(ws + VT_OFF);
  _Float16* Ah    = (_Float16*)(ws + AH_OFF);
  _Float16* Wpt   = (_Float16*)(ws + WPT_OFF);

  // 1) precision conversion / weight transposition
  {
    int n = Mm * Cc;
    cvt_f32_to_f16_kernel<<<(n + 255) / 256, 256, 0, stream>>>(x, Xh, n);
  }
  {
    int n = Cc * Nqkv;
    transpose_to_f16_kernel<<<(n + 255) / 256, 256, 0, stream>>>(Wqkv, Wqkvt, Cc, Nqkv);
  }
  {
    int n = Cc * Cc;
    transpose_to_f16_kernel<<<(n + 255) / 256, 256, 0, stream>>>(Wproj, Wpt, Cc, Cc);
  }

  // 2) QKV projection + head scatter
  gemm_qkv_kernel<<<dim3(Nqkv / 256, Mm / 64), 256, 0, stream>>>(
      Xh, Wqkvt, bqkv, Qh, Kh, Vt);

  // 3) causal flash attention
  attn_kernel<<<dim3(Tt / 16, Bb * Hh), 32, 0, stream>>>(Qh, Kh, Vt, Ah);

  // 4) output projection (fp32 out)
  gemm_proj_kernel<<<dim3(Cc / 256, Mm / 64), 256, 0, stream>>>(
      Ah, Wpt, bproj, out);
}
